// MVN_DDI_Block_15126874816972
// MI455X (gfx1250) — matrix operand order repaired and checked by reading, not compiled, not run on hardware
//
#include <hip/hip_runtime.h>

#define NN 100000
#define GG 1024
#define EINTRA 400000
#define EINTER 300000

typedef __attribute__((ext_vector_type(16))) __bf16 v16bf;
typedef __attribute__((ext_vector_type(8)))  float  v8f;

// order-preserving float->uint encoding for atomicMax-based segment max
__device__ __forceinline__ unsigned fenc(float f) {
  unsigned u = __float_as_uint(f);
  return (u & 0x80000000u) ? ~u : (u | 0x80000000u);
}
__device__ __forceinline__ float fdec(unsigned u) {
  return __uint_as_float((u & 0x80000000u) ? (u & 0x7FFFFFFFu) : ~u);
}
#define ENC_NEG_INF 0x007FFFFFu   // fenc(-inf)

// ------------------------------------------------------------------
// Pack fp32 weight matrix B[K,Nc] into bf16 WMMA B-fragment layout:
//   combo q = (k0/32)*(Nc/16) + tn   (k0 = K-step base, tn = 16-col tile)
//   entry q*512 + lane*16 + j  =  B[(k0 + 16*(lane>>4) + j)*Nc + tn*16 + (lane&15)]
// K*Nc == 8192 for every GEMM in this model -> 16KB per packed matrix.
// ------------------------------------------------------------------
__global__ void pack_b_frag_k(const float* __restrict__ B, __bf16* __restrict__ Bf,
                              int K, int Nc) {
  int tid = blockIdx.x * blockDim.x + threadIdx.x;
  int total = (K * Nc) >> 1;          // packed bf16 pairs
  if (tid >= total) return;
  int e0 = tid << 1;                  // first fragment entry (even j)
  int q  = e0 >> 9;
  int l  = (e0 >> 4) & 31;
  int j0 = e0 & 15;
  int nt = Nc >> 4;
  int k0 = (q / nt) << 5;
  int tn = q % nt;
  int row0 = k0 + ((l >> 4) << 4) + j0;
  int col  = (tn << 4) + (l & 15);
  Bf[e0]     = (__bf16)B[(size_t)row0 * Nc + col];
  Bf[e0 + 1] = (__bf16)B[(size_t)(row0 + 1) * Nc + col];
}

// ------------------------------------------------------------------
// WMMA GEMM: C[M,Nc] = A[M,K] x B[K,Nc]; fp32 A/C, bf16 WMMA core with
// f32 accumulate. B pre-packed in fragment layout, staged to LDS once
// per block. One wave per 16x16 tile (grid-stride). K-step 32.
// ------------------------------------------------------------------
__global__ void gemm_wmma(const float* __restrict__ A, const __bf16* __restrict__ Bf,
                          float* __restrict__ C, int M, int K, int Nc) {
  __shared__ __attribute__((aligned(32))) __bf16 Bs[8192];   // 16KB: full packed B
  const int nchunks = (K * Nc) >> 3;                         // 128-bit chunks
  for (int i = threadIdx.x; i < nchunks; i += blockDim.x)
    ((uint4*)Bs)[i] = ((const uint4*)Bf)[i];                 // ds_store_b128 staging
  __syncthreads();

  const int lane = threadIdx.x & 31;
  const int wave = (blockIdx.x * blockDim.x + threadIdx.x) >> 5;
  const int nwaves = (gridDim.x * blockDim.x) >> 5;
  const int nt = Nc >> 4;
  const int total = (M >> 4) * nt;
  const int half = lane >> 4;      // 0: lanes 0-15, 1: lanes 16-31
  const int l15 = lane & 15;
  for (int t = wave; t < total; t += nwaves) {
    const int row0 = (t / nt) << 4;
    const int col0 = (t % nt) << 4;
    // prefetch next grid-stride tile's A row for this lane (global_prefetch_b8)
    const int tnext = t + nwaves;
    if (tnext < total)
      __builtin_prefetch(A + (size_t)(((tnext / nt) << 4) + l15) * K, 0, 3);
    v8f acc = {};
    const int am = row0 + l15;     // A row handled by this lane
    for (int k0 = 0; k0 < K; k0 += 32) {
      // A 16x32 bf16 layout: lane m=l15, element j -> K=j+16*half
      const float* ap = A + (size_t)am * K + k0 + (half << 4);
      float4 a0 = *(const float4*)(ap);
      float4 a1 = *(const float4*)(ap + 4);
      float4 a2 = *(const float4*)(ap + 8);
      float4 a3 = *(const float4*)(ap + 12);
      v16bf av;
      av[0]=(__bf16)a0.x;  av[1]=(__bf16)a0.y;  av[2]=(__bf16)a0.z;  av[3]=(__bf16)a0.w;
      av[4]=(__bf16)a1.x;  av[5]=(__bf16)a1.y;  av[6]=(__bf16)a1.z;  av[7]=(__bf16)a1.w;
      av[8]=(__bf16)a2.x;  av[9]=(__bf16)a2.y;  av[10]=(__bf16)a2.z; av[11]=(__bf16)a2.w;
      av[12]=(__bf16)a3.x; av[13]=(__bf16)a3.y; av[14]=(__bf16)a3.z; av[15]=(__bf16)a3.w;
      // B fragment: single 32B aligned LDS read (2x ds_load_b128)
      const int q = (k0 >> 5) * nt + (col0 >> 4);
      v16bf bv = *(const v16bf*)&Bs[(q << 9) + (lane << 4)];
      acc = __builtin_amdgcn_wmma_f32_16x16x32_bf16(false, av, false, bv,
                                                    (short)0, acc, false, false);
    }
    // C/D 16x16 f32: vgpr v -> m = v + 8*half, n = l15
#pragma unroll
    for (int v = 0; v < 8; ++v)
      C[(size_t)(row0 + v + (half << 3)) * Nc + col0 + l15] = acc[v];
  }
}

// ------------------------------------------------------------------
// Per-node attention logits: out[n*2+h] = dot(h[n, h*C:(h+1)*C], a[h])
// ------------------------------------------------------------------
__global__ void attn_logits_k(const float* __restrict__ h, const float* __restrict__ a,
                              float* __restrict__ out, int n, int C) {
  int tid = blockIdx.x * blockDim.x + threadIdx.x;
  if (tid >= n * 2) return;
  int node = tid >> 1, hd = tid & 1;
  const float* hp = h + (size_t)node * (2 * C) + hd * C;
  const float* ap = a + hd * C;
  float s = 0.f;
  for (int c = 0; c < C; ++c) s += hp[c] * ap[c];
  out[tid] = s;
}

__global__ void fill_u32_k(unsigned* p, unsigned v, int n) {
  int tid = blockIdx.x * blockDim.x + threadIdx.x;
  if (tid < n) p[tid] = v;
}

// pass 1: per-dst segment max of leaky_relu(es[s]+ed[d]) (incl. L self loops)
__global__ void edge_max_k(const int* __restrict__ src, const int* __restrict__ dst,
                           int E, int L, const float* __restrict__ es,
                           const float* __restrict__ ed, unsigned* __restrict__ mx) {
  int i = blockIdx.x * blockDim.x + threadIdx.x;
  if (i >= E + L) return;
  int s, d;
  if (i < E) { s = src[i]; d = dst[i]; } else { s = d = i - E; }
#pragma unroll
  for (int h = 0; h < 2; ++h) {
    float e = es[s * 2 + h] + ed[d * 2 + h];
    e = e > 0.f ? e : 0.2f * e;
    atomicMax(&mx[d * 2 + h], fenc(e));
  }
}

// pass 2: ex = exp(e - max[d]); accumulate denominator per dst
__global__ void edge_expden_k(const int* __restrict__ src, const int* __restrict__ dst,
                              int E, int L, const float* __restrict__ es,
                              const float* __restrict__ ed, const unsigned* __restrict__ mx,
                              float* __restrict__ den, float* __restrict__ ex) {
  int i = blockIdx.x * blockDim.x + threadIdx.x;
  if (i >= E + L) return;
  int s, d;
  if (i < E) { s = src[i]; d = dst[i]; } else { s = d = i - E; }
#pragma unroll
  for (int h = 0; h < 2; ++h) {
    float e = es[s * 2 + h] + ed[d * 2 + h];
    e = e > 0.f ? e : 0.2f * e;
    float x = __expf(e - fdec(mx[d * 2 + h]));
    ex[i * 2 + h] = x;
    atomicAdd(&den[d * 2 + h], x);
  }
}

// pass 3: out[d, off + c] += ex[i,h] * hs[s, c]; 4 feats per thread (float4 gather)
__global__ void edge_scatter_k(const int* __restrict__ src, const int* __restrict__ dst,
                               int E, int L, const float* __restrict__ ex,
                               const float* __restrict__ hs, int C,
                               float* __restrict__ out, int ldo, int off) {
  int tpe = C >> 1;                       // (2*C)/4 chunks of 4 feats
  long long tid = (long long)blockIdx.x * blockDim.x + threadIdx.x;
  long long tot = (long long)(E + L) * tpe;
  if (tid >= tot) return;
  int i = (int)(tid / tpe), j = (int)(tid % tpe);
  int c0 = j << 2;
  int s, d;
  if (i < E) { s = src[i]; d = dst[i]; } else { s = d = i - E; }
  float a = ex[i * 2 + (c0 >= C ? 1 : 0)];
  float4 v = *(const float4*)(hs + (size_t)s * (2 * C) + c0);
  float* op = out + (size_t)d * ldo + off + c0;
  atomicAdd(op + 0, a * v.x);
  atomicAdd(op + 1, a * v.y);
  atomicAdd(op + 2, a * v.z);
  atomicAdd(op + 3, a * v.w);
}

// pass 4: normalize by denominator, add bias, optional elu (in place)
__global__ void node_norm_k(float* __restrict__ out, int ldo, int off,
                            const float* __restrict__ den, const float* __restrict__ bias,
                            int C, int do_elu, int n) {
  int HC = 2 * C;
  int tid = blockIdx.x * blockDim.x + threadIdx.x;
  if (tid >= n * HC) return;
  int node = tid / HC, c = tid % HC;
  int h = (c >= C) ? 1 : 0;
  size_t idx = (size_t)node * ldo + off + c;
  float v = out[idx] / (den[node * 2 + h] + 1e-16f) + bias[c];
  if (do_elu) v = v > 0.f ? v : __expf(v) - 1.f;
  out[idx] = v;
}

// SAG: agg[d,:] += x[s,:] over raw edges (no self loops)
__global__ void agg_scatter_k(const int* __restrict__ src, const int* __restrict__ dst,
                              int E, const float* __restrict__ x, float* __restrict__ agg) {
  long long tid = (long long)blockIdx.x * blockDim.x + threadIdx.x;
  if (tid >= (long long)E * 32) return;
  int i = (int)(tid >> 5), c0 = ((int)tid & 31) << 2;
  int s = src[i], d = dst[i];
  float4 v = *(const float4*)(x + (size_t)s * 128 + c0);
  float* op = agg + (size_t)d * 128 + c0;
  atomicAdd(op + 0, v.x); atomicAdd(op + 1, v.y);
  atomicAdd(op + 2, v.z); atomicAdd(op + 3, v.w);
}

__global__ void sag_score_k(const float* __restrict__ x, const float* __restrict__ agg,
                            const float* __restrict__ Wrel, const float* __restrict__ brel,
                            const float* __restrict__ Wroot, const int* __restrict__ batch,
                            float* __restrict__ score, unsigned* __restrict__ gmax, int n) {
  int i = blockIdx.x * blockDim.x + threadIdx.x;
  if (i >= n) return;
  const float* xp = x + (size_t)i * 128;
  const float* ap = agg + (size_t)i * 128;
  float s = brel[0];
  for (int c = 0; c < 128; ++c) s += ap[c] * Wrel[c] + xp[c] * Wroot[c];
  score[i] = s;
  atomicMax(&gmax[batch[i]], fenc(s));
}

__global__ void sag_expden_k(const float* __restrict__ score, const int* __restrict__ batch,
                             const unsigned* __restrict__ gmax, float* __restrict__ gden,
                             float* __restrict__ nex, int n) {
  int i = blockIdx.x * blockDim.x + threadIdx.x;
  if (i >= n) return;
  int g = batch[i];
  float m = fdec(gmax[g]);
  if (!(m > -1e38f)) m = 0.f;   // reference: non-finite max -> 0
  float x = __expf(score[i] - m);
  nex[i] = x;
  atomicAdd(&gden[g], x);
}

__global__ void sag_emb_k(const float* __restrict__ x, const float* __restrict__ nex,
                          const int* __restrict__ batch, float* __restrict__ emb, int n) {
  long long tid = (long long)blockIdx.x * blockDim.x + threadIdx.x;
  if (tid >= (long long)n * 32) return;
  int i = (int)(tid >> 5), c0 = ((int)tid & 31) << 2;
  int g = batch[i];
  float a = nex[i];
  float4 v = *(const float4*)(x + (size_t)i * 128 + c0);
  float* op = emb + (size_t)g * 128 + c0;
  atomicAdd(op + 0, a * v.x); atomicAdd(op + 1, a * v.y);
  atomicAdd(op + 2, a * v.z); atomicAdd(op + 3, a * v.w);
}

__global__ void emb_norm_k(float* __restrict__ emb, const float* __restrict__ gden, int g) {
  int tid = blockIdx.x * blockDim.x + threadIdx.x;
  if (tid >= g * 128) return;
  emb[tid] = emb[tid] / (gden[tid >> 7] + 1e-16f);
}

// ------------------------------------------------------------------
extern "C" void kernel_launch(void* const* d_in, const int* in_sizes, int n_in,
                              void* d_out, int out_size, void* d_ws, size_t ws_size,
                              hipStream_t stream) {
  (void)in_sizes; (void)n_in; (void)ws_size;
  const float* x1        = (const float*)d_in[0];
  const int*   ei1       = (const int*)d_in[1];
  const int*   batch1    = (const int*)d_in[2];
  const float* x2        = (const float*)d_in[3];
  const int*   ei2       = (const int*)d_in[4];
  const int*   batch2    = (const int*)d_in[5];
  const int*   bei       = (const int*)d_in[6];
  const float* W_fc      = (const float*)d_in[7];
  const float* a_src_fc  = (const float*)d_in[8];
  const float* a_dst_fc  = (const float*)d_in[9];
  const float* b_fc      = (const float*)d_in[10];
  const float* W_intra   = (const float*)d_in[11];
  const float* a_src_in  = (const float*)d_in[12];
  const float* a_dst_in  = (const float*)d_in[13];
  const float* b_intra   = (const float*)d_in[14];
  const float* W_isrc    = (const float*)d_in[15];
  const float* W_idst    = (const float*)d_in[16];
  const float* a_src_ie  = (const float*)d_in[17];
  const float* a_dst_ie  = (const float*)d_in[18];
  const float* b_inter   = (const float*)d_in[19];
  const float* W_rel     = (const float*)d_in[20];
  const float* b_rel     = (const float*)d_in[21];
  const float* W_root    = (const float*)d_in[22];

  float* d1   = (float*)d_out;
  float* d2   = d1 + (size_t)NN * 128;
  float* emb1 = d2 + (size_t)NN * 128;
  float* emb2 = emb1 + (size_t)GG * 128;

  // workspace carve (256B aligned)
  char* wsp = (char*)d_ws;
  auto carveB = [&](size_t bytes) -> void* {
    void* p = (void*)wsp;
    wsp += ((bytes + 255) / 256) * 256;
    return p;
  };
  auto carve = [&](size_t floats) -> float* { return (float*)carveB(floats * sizeof(float)); };

  float*    e1    = carve((size_t)NN * 128);
  float*    e2    = carve((size_t)NN * 128);
  float*    hsA   = carve((size_t)NN * 128);   // GEMM out / src hs / later agg
  float*    hsB   = carve((size_t)NN * 64);    // inter dst hd
  float*    es    = carve((size_t)NN * 2);
  float*    ed    = carve((size_t)NN * 2);
  unsigned* mx    = (unsigned*)carve((size_t)NN * 2);
  float*    den   = carve((size_t)NN * 2);
  float*    ex    = carve((size_t)(EINTRA + NN) * 2);
  float*    score = carve(NN);
  float*    nex   = carve(NN);
  unsigned* gmax  = (unsigned*)carve(GG);
  float*    gden  = carve(GG);
  __bf16*   Bf_fc    = (__bf16*)carveB(8192 * sizeof(__bf16));  // packed weights
  __bf16*   Bf_intra = (__bf16*)carveB(8192 * sizeof(__bf16));
  __bf16*   Bf_isrc  = (__bf16*)carveB(8192 * sizeof(__bf16));
  __bf16*   Bf_idst  = (__bf16*)carveB(8192 * sizeof(__bf16));

  const int* s1 = ei1;           const int* t1 = ei1 + EINTRA;
  const int* s2 = ei2;           const int* t2 = ei2 + EINTRA;
  const int* bs = bei;           const int* bd = bei + EINTER;

  // pre-pack all GEMM weights into bf16 WMMA fragment layout (16KB each)
  pack_b_frag_k<<<16, 256, 0, stream>>>(W_fc,    Bf_fc,    64, 128);
  pack_b_frag_k<<<16, 256, 0, stream>>>(W_intra, Bf_intra, 128, 64);
  pack_b_frag_k<<<16, 256, 0, stream>>>(W_isrc,  Bf_isrc,  128, 64);
  pack_b_frag_k<<<16, 256, 0, stream>>>(W_idst,  Bf_idst,  128, 64);

  auto gemm = [&](const float* A, const __bf16* Bf, float* C, int M, int K, int Nc) {
    int tiles = (M >> 4) * (Nc >> 4);
    int blocks = (tiles + 7) / 8;          // 8 waves/block, 1 tile/wave (+grid-stride)
    gemm_wmma<<<blocks, 256, 0, stream>>>(A, Bf, C, M, K, Nc);
  };

  auto gat = [&](const float* hs, const float* hd, const float* asrc, const float* adst,
                 const float* bias, const int* src, const int* dst, int E, int C,
                 float* out, int ldo, int off, int elu) {
    int Et = E + NN;
    attn_logits_k<<<(NN * 2 + 255) / 256, 256, 0, stream>>>(hs, asrc, es, NN, C);
    attn_logits_k<<<(NN * 2 + 255) / 256, 256, 0, stream>>>(hd, adst, ed, NN, C);
    fill_u32_k<<<(NN * 2 + 255) / 256, 256, 0, stream>>>(mx, ENC_NEG_INF, NN * 2);
    hipMemsetAsync(den, 0, (size_t)NN * 2 * sizeof(float), stream);
    edge_max_k<<<(Et + 255) / 256, 256, 0, stream>>>(src, dst, E, NN, es, ed, mx);
    edge_expden_k<<<(Et + 255) / 256, 256, 0, stream>>>(src, dst, E, NN, es, ed, mx, den, ex);
    long long sct = (long long)Et * (C >> 1);
    edge_scatter_k<<<(int)((sct + 255) / 256), 256, 0, stream>>>(src, dst, E, NN, ex, hs, C,
                                                                 out, ldo, off);
    node_norm_k<<<(NN * 2 * C + 255) / 256, 256, 0, stream>>>(out, ldo, off, den, bias, C,
                                                              elu, NN);
  };

  hipMemsetAsync(d_out, 0, (size_t)out_size * sizeof(float), stream);

  // ---- shared feature GATConv(64 -> 2x64) + elu, both graphs ----
  hipMemsetAsync(e1, 0, (size_t)NN * 128 * sizeof(float), stream);
  gemm(x1, Bf_fc, hsA, NN, 64, 128);
  gat(hsA, hsA, a_src_fc, a_dst_fc, b_fc, s1, t1, EINTRA, 64, e1, 128, 0, 1);

  hipMemsetAsync(e2, 0, (size_t)NN * 128 * sizeof(float), stream);
  gemm(x2, Bf_fc, hsA, NN, 64, 128);
  gat(hsA, hsA, a_src_fc, a_dst_fc, b_fc, s2, t2, EINTRA, 64, e2, 128, 0, 1);

  // ---- intra GAT (128 -> 2x32) -> d{1,2}_rep[:, 0:64] ----
  gemm(e1, Bf_intra, hsA, NN, 128, 64);
  gat(hsA, hsA, a_src_in, a_dst_in, b_intra, s1, t1, EINTRA, 32, d1, 128, 0, 0);

  gemm(e2, Bf_intra, hsA, NN, 128, 64);
  gat(hsA, hsA, a_src_in, a_dst_in, b_intra, s2, t2, EINTRA, 32, d2, 128, 0, 0);

  // ---- inter GAT t_rep (e1 -> e2 nodes) -> d2_rep[:, 64:128] ----
  gemm(e1, Bf_isrc, hsA, NN, 128, 64);
  gemm(e2, Bf_idst, hsB, NN, 128, 64);
  gat(hsA, hsB, a_src_ie, a_dst_ie, b_inter, bs, bd, EINTER, 32, d2, 128, 64, 0);

  // ---- inter GAT h_rep (e2 -> e1 nodes) -> d1_rep[:, 64:128] ----
  gemm(e2, Bf_isrc, hsA, NN, 128, 64);
  gemm(e1, Bf_idst, hsB, NN, 128, 64);
  gat(hsA, hsB, a_src_ie, a_dst_ie, b_inter, bd, bs, EINTER, 32, d1, 128, 64, 0);

  // ---- SAGPool(min_score=-1) readout, graph 1 ----
  float* agg = hsA;
  hipMemsetAsync(agg, 0, (size_t)NN * 128 * sizeof(float), stream);
  agg_scatter_k<<<(int)(((long long)EINTRA * 32 + 255) / 256), 256, 0, stream>>>(s1, t1, EINTRA,
                                                                                 d1, agg);
  fill_u32_k<<<(GG + 255) / 256, 256, 0, stream>>>(gmax, ENC_NEG_INF, GG);
  hipMemsetAsync(gden, 0, (size_t)GG * sizeof(float), stream);
  sag_score_k<<<(NN + 255) / 256, 256, 0, stream>>>(d1, agg, W_rel, b_rel, W_root, batch1,
                                                    score, gmax, NN);
  sag_expden_k<<<(NN + 255) / 256, 256, 0, stream>>>(score, batch1, gmax, gden, nex, NN);
  sag_emb_k<<<(int)(((long long)NN * 32 + 255) / 256), 256, 0, stream>>>(d1, nex, batch1,
                                                                         emb1, NN);
  emb_norm_k<<<(GG * 128 + 255) / 256, 256, 0, stream>>>(emb1, gden, GG);

  // ---- SAGPool readout, graph 2 ----
  hipMemsetAsync(agg, 0, (size_t)NN * 128 * sizeof(float), stream);
  agg_scatter_k<<<(int)(((long long)EINTRA * 32 + 255) / 256), 256, 0, stream>>>(s2, t2, EINTRA,
                                                                                 d2, agg);
  fill_u32_k<<<(GG + 255) / 256, 256, 0, stream>>>(gmax, ENC_NEG_INF, GG);
  hipMemsetAsync(gden, 0, (size_t)GG * sizeof(float), stream);
  sag_score_k<<<(NN + 255) / 256, 256, 0, stream>>>(d2, agg, W_rel, b_rel, W_root, batch2,
                                                    score, gmax, NN);
  sag_expden_k<<<(NN + 255) / 256, 256, 0, stream>>>(score, batch2, gmax, gden, nex, NN);
  sag_emb_k<<<(int)(((long long)NN * 32 + 255) / 256), 256, 0, stream>>>(d2, nex, batch2,
                                                                         emb2, NN);
  emb_norm_k<<<(GG * 128 + 255) / 256, 256, 0, stream>>>(emb2, gden, GG);
}
